// DeepRNN_16097537426128
// MI455X (gfx1250) — matrix operand order repaired and checked by reading, not compiled, not run on hardware
//
#include <hip/hip_runtime.h>
#include <hip/hip_bf16.h>

// ---------------- problem / tiling constants ----------------
#define T_STEPS 1024
#define BATCH   64
#define HID     512
#define KTOT    512          // K dim of both GEMMs (I == H == 512)
#define SLAB    16           // h-columns owned per workgroup
#define PCOLS   48           // projection columns per WG (3 gates x 16)
#define WROWS   48           // weight rows staged in LDS per matrix
#define NWG     32           // 32 WGs x 16 cols = 512 h-columns
#define BLK     256          // 8 wave32 per WG

typedef __attribute__((ext_vector_type(16))) __bf16         v16bf;
typedef __attribute__((ext_vector_type(16))) unsigned short v16us;
typedef __attribute__((ext_vector_type(8)))  float          v8f;

// fp32 -> bf16 round-to-nearest-even (bit trick)
__device__ inline unsigned short f2bf(float f) {
  unsigned u = __float_as_uint(f);
  u += 0x7FFFu + ((u >> 16) & 1u);
  return (unsigned short)(u >> 16);
}

// A fragment (16x32 bf16, MxK), source already bf16 row-major [.., KTOT].
//  lanes 0-15:  M=lane,    K = [k0..k0+7] ++ [k0+16..k0+23], k0 = 32*ks
//  lanes 16-31: M=lane-16, same but k0 = 32*ks + 8
// -> two 16B contiguous loads, zero VALU.
__device__ inline v16bf load_a_frag_bf(const unsigned short* __restrict__ src,
                                       int mrow0, int ks, int lane) {
  const int m  = mrow0 + (lane & 15);
  const int k0 = ks * 32 + ((lane & 16) ? 8 : 0);
  const unsigned short* p = src + (size_t)m * KTOT + k0;
  union { v16us v; uint4 q[2]; } u;
  u.q[0] = *(const uint4*)p;          // K = k0   .. k0+7
  u.q[1] = *(const uint4*)(p + 16);   // K = k0+16.. k0+23
  return __builtin_bit_cast(v16bf, u.v);
}

// Fallback A fragment from fp32 source: convert in registers.
__device__ inline v16bf load_a_frag_cvt(const float* __restrict__ src,
                                        int mrow0, int ks, int lane) {
  const int m  = mrow0 + (lane & 15);
  const int k0 = ks * 32 + ((lane & 16) ? 8 : 0);
  const float4* p = (const float4*)(src + (size_t)m * KTOT + k0);
  float4 a0 = p[0], a1 = p[1];
  float4 b0 = p[4], b1 = p[5];
  v16us r;
  r[0]  = f2bf(a0.x); r[1]  = f2bf(a0.y); r[2]  = f2bf(a0.z); r[3]  = f2bf(a0.w);
  r[4]  = f2bf(a1.x); r[5]  = f2bf(a1.y); r[6]  = f2bf(a1.z); r[7]  = f2bf(a1.w);
  r[8]  = f2bf(b0.x); r[9]  = f2bf(b0.y); r[10] = f2bf(b0.z); r[11] = f2bf(b0.w);
  r[12] = f2bf(b1.x); r[13] = f2bf(b1.y); r[14] = f2bf(b1.z); r[15] = f2bf(b1.w);
  return __builtin_bit_cast(v16bf, r);
}

// B fragment (32x16 bf16, KxN): lane%16 = N (weight row); K = [k0..k0+7] ++
// [k0+16..k0+23], k0 = 32*ks + 8*(lane/16) (2x ds_load_b128).
__device__ inline v16bf load_b_frag(const unsigned short* __restrict__ w,
                                    int row, int ks, int lane) {
  const int k0 = ks * 32 + ((lane & 16) ? 8 : 0);
  const unsigned short* p = w + (size_t)row * KTOT + k0;
  union { v16us v; uint4 q[2]; } u;
  u.q[0] = *(const uint4*)p;
  u.q[1] = *(const uint4*)(p + 16);
  return __builtin_bit_cast(v16bf, u.v);
}

// device-scope grid barrier (persistent kernel, NWG co-resident WGs)
__device__ inline void grid_barrier(unsigned* cnt, unsigned* gen, int target) {
  __builtin_amdgcn_fence(__ATOMIC_RELEASE, "agent");   // global_wb
  __syncthreads();
  if (threadIdx.x == 0) {
    unsigned prev = __hip_atomic_fetch_add(cnt, 1u, __ATOMIC_ACQ_REL,
                                           __HIP_MEMORY_SCOPE_AGENT);
    if (prev == (unsigned)(NWG - 1)) {
      __hip_atomic_store(cnt, 0u, __ATOMIC_RELAXED, __HIP_MEMORY_SCOPE_AGENT);
      __hip_atomic_store(gen, (unsigned)target, __ATOMIC_RELEASE,
                         __HIP_MEMORY_SCOPE_AGENT);
    } else {
      while (__hip_atomic_load(gen, __ATOMIC_ACQUIRE,
                               __HIP_MEMORY_SCOPE_AGENT) < (unsigned)target) {
        __builtin_amdgcn_s_sleep(2);
      }
    }
  }
  __syncthreads();
  __builtin_amdgcn_fence(__ATOMIC_ACQUIRE, "agent");   // global_inv
}

__global__ void init_bar(unsigned* cnt, unsigned* gen) {
  if (threadIdx.x == 0) { *cnt = 0u; *gen = 0u; }
}

// bulk fp32 -> bf16 conversion (layer-1 input), vectorized x4
__global__ void xcvt(const float* __restrict__ x, unsigned short* __restrict__ xb,
                     size_t n4) {
  size_t i = (size_t)blockIdx.x * blockDim.x + threadIdx.x;
  size_t stride = (size_t)gridDim.x * blockDim.x;
  for (; i < n4; i += stride) {
    float4 v = ((const float4*)x)[i];
    ushort4 o;
    o.x = f2bf(v.x); o.y = f2bf(v.y); o.z = f2bf(v.z); o.w = f2bf(v.w);
    ((ushort4*)xb)[i] = o;
  }
}

template <bool PRE>
__global__ __launch_bounds__(BLK)
void gru_persistent(const float* __restrict__ x,
                    const float* __restrict__ wih0, const float* __restrict__ whh0,
                    const float* __restrict__ bih0, const float* __restrict__ bhh0,
                    const float* __restrict__ wih1, const float* __restrict__ whh1,
                    const float* __restrict__ bih1, const float* __restrict__ bhh1,
                    float* __restrict__ out,
                    unsigned short* __restrict__ hbf,   // 2 x [64][512] bf16 ping-pong
                    unsigned short* __restrict__ x1bf,  // [T][64][512] bf16 (PRE only)
                    unsigned short* __restrict__ x2bf,  // [T][64][512] bf16 (PRE only)
                    unsigned* __restrict__ bar_cnt,
                    unsigned* __restrict__ bar_gen) {
  extern __shared__ char smem[];
  unsigned short* wih_s = (unsigned short*)smem;            // 48*512 bf16
  unsigned short* whh_s = wih_s + WROWS * KTOT;             // 48*512 bf16
  float* xp_st = (float*)(whh_s + WROWS * KTOT);            // [64][48] fp32
  float* hp_st = xp_st + BATCH * PCOLS;                     // [64][48] fp32
  float* bih_s = hp_st + BATCH * PCOLS;                     // [48]
  float* bhh_s = bih_s + PCOLS;                             // [48]

  const int tid   = threadIdx.x;
  const int lane  = tid & 31;
  const int wid   = tid >> 5;
  const int gemm  = wid >> 2;          // 0: input proj, 1: recurrent proj
  const int mtile = wid & 3;           // which 16-row band of B=64
  const int slab  = blockIdx.x * SLAB; // first h-column owned by this WG
  const size_t TBH = (size_t)T_STEPS * BATCH * HID;

  unsigned short* hbfA = hbf;
  unsigned short* hbfB = hbf + BATCH * HID;
  int epoch = 0;

  for (int layer = 0; layer < 2; ++layer) {
    const float* wih = layer ? wih1 : wih0;
    const float* whh = layer ? whh1 : whh0;
    const float* bih = layer ? bih1 : bih0;
    const float* bhh = layer ? bhh1 : bhh0;
    const float* xf32base = layer ? (out + TBH) : x;     // fp32 input (fallback)
    unsigned short* xbfbase = layer ? x2bf : x1bf;       // bf16 input (PRE)
    float* o1 = layer ? out : (out + TBH);               // h2 slot0 | h1 slot
    float* o2 = layer ? (out + 2 * TBH) : nullptr;       // h2 slot2

    // ---- stage this WG's weight rows (bf16) + biases into LDS ----
    for (int idx = tid; idx < WROWS * KTOT; idx += BLK) {
      int rl = idx >> 9;               // local row 0..47
      int k  = idx & (KTOT - 1);
      int g  = rl >> 4;                // gate 0..2 (r,z,n)
      int j  = rl & 15;
      int grow = g * HID + slab + j;   // row of [3H, K] weight
      wih_s[idx] = f2bf(wih[(size_t)grow * KTOT + k]);
      whh_s[idx] = f2bf(whh[(size_t)grow * KTOT + k]);
    }
    for (int idx = tid; idx < PCOLS; idx += BLK) {
      int g = idx >> 4, j = idx & 15;
      bih_s[idx] = bih[g * HID + slab + j];
      bhh_s[idx] = bhh[g * HID + slab + j];
    }
    // zero the bf16 h-state slab read at t=0
    for (int i = tid; i < BATCH * SLAB; i += BLK) {
      int m = i >> 4, j = i & 15;
      hbfA[(size_t)m * HID + slab + j] = 0;
    }
    grid_barrier(bar_cnt, bar_gen, ++epoch);

    for (int t = 0; t < T_STEPS; ++t) {
      unsigned short* hbf_cur = (t & 1) ? hbfB : hbfA;
      unsigned short* hbf_nxt = (t & 1) ? hbfA : hbfB;

      // ---- WMMA phase: waves 0-3 -> xp, waves 4-7 -> hp ----
      const unsigned short* wmat = gemm ? whh_s : wih_s;
      float* stage = gemm ? hp_st : xp_st;
      const unsigned short* abf =
          gemm ? hbf_cur : (PRE ? xbfbase + (size_t)t * BATCH * HID : nullptr);
      const float* af32 = xf32base + (size_t)t * BATCH * HID;

      v8f acc0 = {}, acc1 = {}, acc2 = {};
      const int brow = lane & 15;
#pragma unroll
      for (int ks = 0; ks < KTOT / 32; ++ks) {
        v16bf a;
        if (PRE || gemm) a = load_a_frag_bf(abf, mtile * 16, ks, lane);
        else             a = load_a_frag_cvt(af32, mtile * 16, ks, lane);
        v16bf br = load_b_frag(wmat,  0 + brow, ks, lane);
        v16bf bz = load_b_frag(wmat, 16 + brow, ks, lane);
        v16bf bn = load_b_frag(wmat, 32 + brow, ks, lane);
        acc0 = __builtin_amdgcn_wmma_f32_16x16x32_bf16(false, a, false, br,
                                                       (short)0, acc0, false, false);
        acc1 = __builtin_amdgcn_wmma_f32_16x16x32_bf16(false, a, false, bz,
                                                       (short)0, acc1, false, false);
        acc2 = __builtin_amdgcn_wmma_f32_16x16x32_bf16(false, a, false, bn,
                                                       (short)0, acc2, false, false);
      }
      // spill accumulators to LDS staging (C/D layout: M=v | M=v+8)
      {
        const int mr = mtile * 16 + ((lane & 16) ? 8 : 0);
        const int cl = lane & 15;
#pragma unroll
        for (int v = 0; v < 8; ++v) {
          stage[(mr + v) * PCOLS +      cl] = acc0[v];
          stage[(mr + v) * PCOLS + 16 + cl] = acc1[v];
          stage[(mr + v) * PCOLS + 32 + cl] = acc2[v];
        }
      }
      __syncthreads();

      // ---- fused gate phase: 64x16 elements, 4 per thread ----
#pragma unroll
      for (int q = 0; q < 4; ++q) {
        int e = tid + q * BLK;         // 0..1023
        int m = e >> 4, j = e & 15;
        float xr = xp_st[m * PCOLS + j]      + bih_s[j];
        float xz = xp_st[m * PCOLS + 16 + j] + bih_s[16 + j];
        float xn = xp_st[m * PCOLS + 32 + j] + bih_s[32 + j];
        float hr = hp_st[m * PCOLS + j]      + bhh_s[j];
        float hz = hp_st[m * PCOLS + 16 + j] + bhh_s[16 + j];
        float hn = hp_st[m * PCOLS + 32 + j] + bhh_s[32 + j];
        float r = 1.f / (1.f + __expf(-(xr + hr)));
        float z = 1.f / (1.f + __expf(-(xz + hz)));
        float n = tanhf(xn + r * hn);
        size_t hidx = (size_t)m * HID + slab + j;
        size_t oidx = ((size_t)t * BATCH + m) * HID + slab + j;
        // h_prev: fp32 value this WG wrote to the output at step t-1
        float hp = (t == 0) ? 0.f : o1[oidx - (size_t)BATCH * HID];
        float h  = (1.f - z) * n + z * hp;
        h = (fabsf(h) >= 0.02f) ? h : 0.f;   // theta mask
        unsigned short hb = f2bf(h);
        hbf_nxt[hidx] = hb;                  // bf16 state for next recurrent GEMM
        if (PRE && layer == 0) x2bf[oidx] = hb;  // bf16 h1 = layer-2 GEMM input
        o1[oidx] = h;
        if (o2) o2[oidx] = h;
      }

      // prefetch next step's input tile (global_prefetch_b8)
      if (t + 1 < T_STEPS) {
        if (PRE) {
          const char* nx = (const char*)(xbfbase + (size_t)(t + 1) * BATCH * HID);
          __builtin_prefetch(nx + (size_t)tid * 256, 0, 1);
        } else {
          const char* nx = (const char*)(xf32base + (size_t)(t + 1) * BATCH * HID);
          __builtin_prefetch(nx + (size_t)tid * 512, 0, 1);
        }
      }
      grid_barrier(bar_cnt, bar_gen, ++epoch);
    }
  }
}

extern "C" void kernel_launch(void* const* d_in, const int* in_sizes, int n_in,
                              void* d_out, int out_size, void* d_ws, size_t ws_size,
                              hipStream_t stream) {
  const float* x    = (const float*)d_in[0];
  const float* wih0 = (const float*)d_in[1];
  const float* whh0 = (const float*)d_in[2];
  const float* bih0 = (const float*)d_in[3];
  const float* bhh0 = (const float*)d_in[4];
  const float* wih1 = (const float*)d_in[5];
  const float* whh1 = (const float*)d_in[6];
  const float* bih1 = (const float*)d_in[7];
  const float* bhh1 = (const float*)d_in[8];
  float* out = (float*)d_out;

  const size_t TBH = (size_t)T_STEPS * BATCH * HID;

  unsigned* bar = (unsigned*)d_ws;                        // cnt @ +0, gen @ +128B
  size_t off_h  = 256;
  size_t off_x1 = off_h + (size_t)2 * BATCH * HID * 2;    // 2 bf16 h buffers
  size_t off_x2 = off_x1 + TBH * 2;                       // bf16 x (layer 1)
  size_t need   = off_x2 + TBH * 2;                       // bf16 h1 (layer 2 in)

  unsigned short* hbf  = (unsigned short*)((char*)d_ws + off_h);
  unsigned short* x1bf = (unsigned short*)((char*)d_ws + off_x1);
  unsigned short* x2bf = (unsigned short*)((char*)d_ws + off_x2);

  const bool pre = (ws_size >= need);

  init_bar<<<1, 64, 0, stream>>>(bar, bar + 32);

  size_t smem = (size_t)(2 * WROWS * KTOT) * sizeof(unsigned short)  // weights
              + (size_t)(2 * BATCH * PCOLS) * sizeof(float)          // staging
              + (size_t)(2 * PCOLS) * sizeof(float);                 // biases

  if (pre) {
    xcvt<<<2048, BLK, 0, stream>>>(x, x1bf, TBH / 4);
    gru_persistent<true><<<NWG, BLK, smem, stream>>>(
        x, wih0, whh0, bih0, bhh0, wih1, whh1, bih1, bhh1,
        out, hbf, x1bf, x2bf, bar, bar + 32);
  } else {
    gru_persistent<false><<<NWG, BLK, smem, stream>>>(
        x, wih0, whh0, bih0, bhh0, wih1, whh1, bih1, bhh1,
        out, hbf, nullptr, nullptr, bar, bar + 32);
  }
}
